// Img2Vec_SNN_46969762349239
// MI455X (gfx1250) — matrix operand order repaired and checked by reading, not compiled, not run on hardware
//
#include <hip/hip_runtime.h>

// ---------------------------------------------------------------------------
// Problem constants (from reference): B=4096, D=2048, H=256, C=2, T=100
// ---------------------------------------------------------------------------
#define BB 4096
#define DD 2048
#define HH 256
#define TT 100

typedef float  v8f   __attribute__((ext_vector_type(8)));
typedef __bf16 v8bf  __attribute__((ext_vector_type(8)));
typedef __bf16 v16bf __attribute__((ext_vector_type(16)));

// ---------------------------------------------------------------------------
// K0: split W1 [256 x 2048] f32 -> (W1h, W1l) bf16 pairs: W1 ~= W1h + W1l.
// ---------------------------------------------------------------------------
__global__ void k0_split_w1(const float* __restrict__ W1,
                            __bf16* __restrict__ W1h,
                            __bf16* __restrict__ W1l) {
  const int i = blockIdx.x * 256 + threadIdx.x;  // 512K elements
  const float v  = W1[i];
  const __bf16 h = (__bf16)v;
  W1h[i] = h;
  W1l[i] = (__bf16)(v - (float)h);
}

// ---------------------------------------------------------------------------
// K1: cur1[4096 x 256] = x @ W1^T + b1 via bf16 hi/lo-split WMMA 16x16x32
// (x_hi*W_hi + x_hi*W_lo + x_lo*W_hi, fp32 accumulate -> fp32-class accuracy).
//
// Block = 256 threads (8 waves), covers M=32 x N=256.
//   wave w: m-half = w>>2 (16 rows), n-slab = (w&3)*64 (4 N-tiles, 4 accs).
// x chunk [32 x 64] staged through LDS, split to bf16 hi/lo on the fly.
// LDS row stride 72 bf16 -> bank = (36m + c) % 64: conflict-free A reads.
// B fragment needs K-contiguous per lane -> row-major W1h/W1l is already
// the correct layout (no transpose kernel needed).
// ---------------------------------------------------------------------------
__global__ void k1_gemm_wmma_bf16x3(const float* __restrict__ x,
                                    const __bf16* __restrict__ W1h,
                                    const __bf16* __restrict__ W1l,
                                    const float* __restrict__ b1,
                                    float* __restrict__ cur1) {
  __shared__ __bf16 lds_ah[32 * 72];
  __shared__ __bf16 lds_al[32 * 72];

  const int m0     = blockIdx.x * 32;
  const int lane   = threadIdx.x & 31;
  const int wave   = threadIdx.x >> 5;
  const int n_base = (wave & 3) * 64;      // 4 n-slabs of 64
  const int mloc   = (wave >> 2) * 16;     // 2 m-halves of 16
  const int ln16   = lane & 15;            // M index (A) / N index (B)
  const int half   = lane >> 4;            // 0 or 1
  const int arow   = (mloc + ln16) * 72;   // this lane's A row in LDS

  v8f acc[4] = {{}, {}, {}, {}};

  for (int kk = 0; kk < DD; kk += 64) {
    // ---- stage + split x chunk [32 x 64]: 8 floats per thread ----
    {
      const int t = threadIdx.x;
      const int r = t >> 3;              // 0..31
      const int c = (t & 7) << 3;        // 0,8,...,56
      const float4* src =
          reinterpret_cast<const float4*>(&x[(m0 + r) * DD + kk + c]);
      float4 p0 = src[0];
      float4 p1 = src[1];
      float vals[8] = {p0.x, p0.y, p0.z, p0.w, p1.x, p1.y, p1.z, p1.w};
      v8bf hv, lv;
#pragma unroll
      for (int i = 0; i < 8; ++i) {
        const __bf16 h = (__bf16)vals[i];
        hv[i] = h;
        lv[i] = (__bf16)(vals[i] - (float)h);
      }
      *reinterpret_cast<v8bf*>(&lds_ah[r * 72 + c]) = hv;
      *reinterpret_cast<v8bf*>(&lds_al[r * 72 + c]) = lv;
      if (kk + 64 < DD)  // pull next chunk toward L2 (global_prefetch_b8)
        __builtin_prefetch(&x[(m0 + r) * DD + kk + 64 + c], 0, 1);
    }
    __syncthreads();

    // ---- two K=32 WMMA groups per chunk ----
#pragma unroll
    for (int g = 0; g < 64; g += 32) {
      // A fragments: runs K = g + half*8 + [0..7] and +16 (ISA 16-bit A map)
      const int ko = g + half * 8;
      v8bf ah0 = *reinterpret_cast<const v8bf*>(&lds_ah[arow + ko]);
      v8bf ah1 = *reinterpret_cast<const v8bf*>(&lds_ah[arow + ko + 16]);
      v8bf al0 = *reinterpret_cast<const v8bf*>(&lds_al[arow + ko]);
      v8bf al1 = *reinterpret_cast<const v8bf*>(&lds_al[arow + ko + 16]);
      v16bf ah = __builtin_shufflevector(ah0, ah1, 0, 1, 2, 3, 4, 5, 6, 7, 8,
                                         9, 10, 11, 12, 13, 14, 15);
      v16bf al = __builtin_shufflevector(al0, al1, 0, 1, 2, 3, 4, 5, 6, 7, 8,
                                         9, 10, 11, 12, 13, 14, 15);

      // B fragments: lane holds K = kcol..kcol+7 and kcol+16..kcol+23 for
      // column n (same K map as A; row-major W1h/W1l)
      const int kcol = kk + g + half * 8;
#pragma unroll
      for (int j = 0; j < 4; ++j) {
        const int n = n_base + j * 16 + ln16;
        const __bf16* bhp = &W1h[(size_t)n * DD + kcol];
        const __bf16* blp = &W1l[(size_t)n * DD + kcol];
        v8bf bh0 = *reinterpret_cast<const v8bf*>(bhp);
        v8bf bh1 = *reinterpret_cast<const v8bf*>(bhp + 16);
        v8bf bl0 = *reinterpret_cast<const v8bf*>(blp);
        v8bf bl1 = *reinterpret_cast<const v8bf*>(blp + 16);
        v16bf bh = __builtin_shufflevector(bh0, bh1, 0, 1, 2, 3, 4, 5, 6, 7,
                                           8, 9, 10, 11, 12, 13, 14, 15);
        v16bf bl = __builtin_shufflevector(bl0, bl1, 0, 1, 2, 3, 4, 5, 6, 7,
                                           8, 9, 10, 11, 12, 13, 14, 15);

        acc[j] = __builtin_amdgcn_wmma_f32_16x16x32_bf16(
            false, ah, false, bh, (short)0, acc[j], false, false);
        acc[j] = __builtin_amdgcn_wmma_f32_16x16x32_bf16(
            false, ah, false, bl, (short)0, acc[j], false, false);
        acc[j] = __builtin_amdgcn_wmma_f32_16x16x32_bf16(
            false, al, false, bh, (short)0, acc[j], false, false);
      }
    }
    __syncthreads();
  }

  // ---- epilogue: D layout: VGPR r -> M = r + 8*half, N = lane&15 ----
  const int mrow = half * 8;
#pragma unroll
  for (int j = 0; j < 4; ++j) {
    const int n    = n_base + j * 16 + ln16;
    const float bv = b1[n];
#pragma unroll
    for (int r = 0; r < 8; ++r)
      cur1[(size_t)(m0 + mloc + mrow + r) * HH + n] = acc[j][r] + bv;
  }
}

// ---------------------------------------------------------------------------
// K2: fused SNN dynamics. One wave32 per batch row b; lane owns 8 hidden
// units (h = lane + 32*i). All 100 steps in registers; layer-2 reduction via
// __shfl_xor butterfly (all lanes end with the full sum -> redundant but
// divergence-free mem2 tracking). Lane 0 writes acc/T.
// ---------------------------------------------------------------------------
__global__ void k2_snn_dynamics(const float* __restrict__ cur1,
                                const float* __restrict__ W2,
                                const float* __restrict__ b2,
                                float* __restrict__ out) {
  const int lane = threadIdx.x & 31;
  const int wv   = threadIdx.x >> 5;
  const int b    = blockIdx.x * 8 + wv;  // 512 blocks * 8 waves = 4096 rows

  float c1[8], w20[8], w21[8], m1[8];
#pragma unroll
  for (int i = 0; i < 8; ++i) {
    const int h = lane + 32 * i;
    c1[i]  = cur1[b * HH + h];
    w20[i] = W2[h];          // W2[0][h]
    w21[i] = W2[HH + h];     // W2[1][h]
    m1[i]  = 0.0f;
  }
  const float b2a = b2[0];
  const float b2b = b2[1];

  float mem2a = 0.0f, mem2b = 0.0f, acca = 0.0f, accb = 0.0f;

  for (int t = 0; t < TT; ++t) {
    float s0 = 0.0f, s1 = 0.0f;
#pragma unroll
    for (int i = 0; i < 8; ++i) {
      // reset from PRE-update membrane (subtract mechanism, threshold = 1)
      const float reset = (m1[i] > 1.0f) ? 1.0f : 0.0f;
      m1[i] = 0.9f * m1[i] + c1[i] - reset;
      const float spk = (m1[i] > 1.0f) ? 1.0f : 0.0f;  // mem1n - 1 > 0
      s0 = __builtin_fmaf(spk, w20[i], s0);
      s1 = __builtin_fmaf(spk, w21[i], s1);
    }
    // wave32 butterfly: every lane gets the full 256-wide sums
#pragma unroll
    for (int off = 16; off > 0; off >>= 1) {
      s0 += __shfl_xor(s0, off);
      s1 += __shfl_xor(s1, off);
    }
    const float cur2a = s0 + b2a;
    const float cur2b = s1 + b2b;
    const float r2a = (mem2a > 1.0f) ? 1.0f : 0.0f;
    const float r2b = (mem2b > 1.0f) ? 1.0f : 0.0f;
    mem2a = 0.9f * mem2a + cur2a - r2a;
    mem2b = 0.9f * mem2b + cur2b - r2b;
    acca += (mem2a > 1.0f) ? 1.0f : 0.0f;
    accb += (mem2b > 1.0f) ? 1.0f : 0.0f;
  }

  if (lane == 0) {
    out[b * 2 + 0] = acca * (1.0f / TT);
    out[b * 2 + 1] = accb * (1.0f / TT);
  }
}

// ---------------------------------------------------------------------------
// launcher
// ---------------------------------------------------------------------------
extern "C" void kernel_launch(void* const* d_in, const int* in_sizes, int n_in,
                              void* d_out, int out_size, void* d_ws,
                              size_t ws_size, hipStream_t stream) {
  (void)in_sizes; (void)n_in; (void)out_size; (void)ws_size;
  const float* x  = (const float*)d_in[0];  // [4096, 2048]
  const float* W1 = (const float*)d_in[1];  // [256, 2048]
  const float* b1 = (const float*)d_in[2];  // [256]
  const float* W2 = (const float*)d_in[3];  // [2, 256]
  const float* b2 = (const float*)d_in[4];  // [2]
  float* out = (float*)d_out;               // [4096, 2]

  // workspace: W1h (1MB) | W1l (1MB) | cur1 (4MB)
  __bf16* w1h  = (__bf16*)d_ws;
  __bf16* w1l  = w1h + (size_t)HH * DD;
  float*  cur1 = (float*)(w1l + (size_t)HH * DD);

  k0_split_w1<<<dim3((HH * DD) / 256), dim3(256), 0, stream>>>(W1, w1h, w1l);
  k1_gemm_wmma_bf16x3<<<dim3(BB / 32), dim3(256), 0, stream>>>(x, w1h, w1l,
                                                               b1, cur1);
  k2_snn_dynamics<<<dim3(BB / 8), dim3(256), 0, stream>>>(cur1, W2, b2, out);
}
